// Block_62311385530381
// MI455X (gfx1250) — matrix operand order repaired and checked by reading, not compiled, not run on hardware
//
#include <hip/hip_runtime.h>

#define DMODEL 2048
#define NTOK   16384   // B*S = 4*4096
#define DHEAD  128

typedef __attribute__((ext_vector_type(16))) __bf16 v16bf;
typedef __attribute__((ext_vector_type(8)))  float  v8f;
typedef unsigned int u32x4 __attribute__((ext_vector_type(4)));
typedef int          i32x8 __attribute__((ext_vector_type(8)));
typedef int          i32x4 __attribute__((ext_vector_type(4)));

union Frag16 { v16bf v; uint4 q[2]; unsigned short u[16]; };

static __device__ __forceinline__ unsigned short f2bf(float f) {
  unsigned u = __float_as_uint(f);
  return (unsigned short)((u + 0x7FFFu + ((u >> 16) & 1u)) >> 16);  // RNE
}
static __device__ __forceinline__ unsigned pack2(float a, float b) {
  return (unsigned)f2bf(a) | ((unsigned)f2bf(b) << 16);
}
static __device__ __forceinline__ v8f wmma_bf16(v16bf a, v16bf b, v8f c) {
  return __builtin_amdgcn_wmma_f32_16x16x32_bf16(false, a, false, b, (short)0, c,
                                                 false, false);
}
static __device__ __forceinline__ float gelu_exact(float x) {
  return 0.5f * x * (1.0f + erff(x * 0.70710678118654752f));
}

// ---------------------------------------------------------------------------
// TDM: DMA one 128(rows) x 32(elem) bf16 tile (row stride DMODEL elems) into
// LDS at lds_off. 2-D D# per CDNA5 ISA; groups 2/3 = 0.
// Toolchain exposes the 6-arg builtin (extra i32x8 group before cpol).
// ---------------------------------------------------------------------------
static __device__ __forceinline__ void tdm_load_tile_128x32(
    const unsigned short* gsrc, unsigned lds_off)
{
  const unsigned long long ga = (unsigned long long)(size_t)gsrc;
  const u32x4 g0 = {
    1u,                                               // count=1 (valid user D#)
    lds_off,                                          // lds_addr (bytes)
    (unsigned)ga,                                     // global_addr[31:0]
    ((unsigned)(ga >> 32) & 0x01FFFFFFu) | (2u << 30) // addr[56:32] | type=2
  };
  const i32x8 g1 = {
    (int)(1u << 16),                    // workgroup_mask=0, data_size=1 (2B)
    (int)((DMODEL & 0xFFFFu) << 16),    // tensor_dim0[15:0] = 2048
    (int)((DMODEL >> 16) | (128u << 16)),// dim0 hi | tensor_dim1[15:0]=128
    (int)(32u << 16),                   // dim1 hi=0 | tile_dim0=32
    128,                                // tile_dim1=128 | tile_dim2=0
    DMODEL,                             // tensor_dim0_stride lo = 2048
    0, 0                                // stride0 hi | dim1_stride = 0
  };
  const i32x4 gz4 = {0, 0, 0, 0};
  const i32x8 gz8 = {0, 0, 0, 0, 0, 0, 0, 0};
  __builtin_amdgcn_tensor_load_to_lds(g0, g1, gz4, gz4, gz8, 0);
}

// ---------------------------------------------------------------------------
// fp32 -> bf16 streaming convert (8 elems/thread)
// ---------------------------------------------------------------------------
__global__ __launch_bounds__(256) void f32_to_bf16(
    const float* __restrict__ in, unsigned short* __restrict__ out)
{
  const size_t i = ((size_t)blockIdx.x * 256 + threadIdx.x) * 8;
  const float4* p = (const float4*)(in + i);
  float4 a = p[0], b = p[1];
  *(uint4*)(out + i) = make_uint4(pack2(a.x, a.y), pack2(a.z, a.w),
                                  pack2(b.x, b.y), pack2(b.z, b.w));
}

// ---------------------------------------------------------------------------
// W[K][N] fp32 -> WT[N][K] bf16  (32x32 LDS tile transpose)
// ---------------------------------------------------------------------------
__global__ __launch_bounds__(256) void w_transpose_bf16(
    const float* __restrict__ W, unsigned short* __restrict__ WT)
{
  __shared__ float t[32][33];
  const int c  = threadIdx.x & 31;
  const int r0 = threadIdx.x >> 5;          // 0..7
  const int n0 = blockIdx.x * 32;
  const int k0 = blockIdx.y * 32;
#pragma unroll
  for (int i = 0; i < 4; ++i) {
    const int r = r0 + i * 8;
    t[r][c] = W[(size_t)(k0 + r) * DMODEL + n0 + c];
  }
  __syncthreads();
#pragma unroll
  for (int i = 0; i < 4; ++i) {
    const int r = r0 + i * 8;
    WT[(size_t)(n0 + r) * DMODEL + k0 + c] = f2bf(t[c][r]);
  }
}

// ---------------------------------------------------------------------------
// C = act(A[16384,2048]bf16 @ WT[N][K]bf16^T + bias). 128x128 tile, BK=32,
// 8 waves x (2x4) WMMA tiles. Tiles DMA'd by TDM, double buffered in LDS.
// ---------------------------------------------------------------------------
__global__ __launch_bounds__(256) void gemm_bf16_tdm(
    const unsigned short* __restrict__ A, const unsigned short* __restrict__ WT,
    const float* __restrict__ bias, float* __restrict__ outF,
    unsigned short* __restrict__ outB, int gelu)
{
  __shared__ __align__(16) unsigned short As[2][128][32];  // [m][k]
  __shared__ __align__(16) unsigned short Bs[2][128][32];  // [n][k]

  const int tid  = threadIdx.x;
  const int lane = tid & 31;
  const int wid  = tid >> 5;
  const int lr   = lane & 15;
  const int half = lane >> 4;
  const int wm   = wid & 3;
  const int wn   = wid >> 2;
  const int bM   = blockIdx.y * 128;
  const int bN   = blockIdx.x * 128;

  const v8f zero8 = {0.f, 0.f, 0.f, 0.f, 0.f, 0.f, 0.f, 0.f};
  v8f acc[2][4];
#pragma unroll
  for (int i = 0; i < 2; ++i)
#pragma unroll
    for (int j = 0; j < 4; ++j) acc[i][j] = zero8;

  const unsigned ldsA[2] = {(unsigned)(size_t)&As[0][0][0],
                            (unsigned)(size_t)&As[1][0][0]};
  const unsigned ldsB[2] = {(unsigned)(size_t)&Bs[0][0][0],
                            (unsigned)(size_t)&Bs[1][0][0]};
  const unsigned short* gA = A  + (size_t)bM * DMODEL;
  const unsigned short* gB = WT + (size_t)bN * DMODEL;

  if (wid == 0) {                       // one wave drives the TDM
    tdm_load_tile_128x32(gA, ldsA[0]);
    tdm_load_tile_128x32(gB, ldsB[0]);
    __builtin_amdgcn_s_wait_tensorcnt(0);
  }
  __syncthreads();

  for (int k0 = 0; k0 < DMODEL; k0 += 32) {
    const int buf = (k0 >> 5) & 1;
    if (wid == 0 && (k0 + 32) < DMODEL) {    // prefetch next tiles via TDM
      tdm_load_tile_128x32(gA + k0 + 32, ldsA[buf ^ 1]);
      tdm_load_tile_128x32(gB + k0 + 32, ldsB[buf ^ 1]);
    }

    // fragments per ISA 16-bit A 16x32 / B 32x16 lane layouts
    Frag16 fa[2], fb[4];
#pragma unroll
    for (int mt = 0; mt < 2; ++mt) {
      const int row = (wm * 2 + mt) * 16 + lr;
      fa[mt].q[0] = *(const uint4*)&As[buf][row][half * 8];
      fa[mt].q[1] = *(const uint4*)&As[buf][row][16 + half * 8];
    }
#pragma unroll
    for (int nt = 0; nt < 4; ++nt) {
      const int col = (wn * 4 + nt) * 16 + lr;
      fb[nt].q[0] = *(const uint4*)&Bs[buf][col][half * 8];
      fb[nt].q[1] = *(const uint4*)&Bs[buf][col][16 + half * 8];
    }
#pragma unroll
    for (int mt = 0; mt < 2; ++mt)
#pragma unroll
      for (int nt = 0; nt < 4; ++nt)
        acc[mt][nt] = wmma_bf16(fa[mt].v, fb[nt].v, acc[mt][nt]);

    if (wid == 0) __builtin_amdgcn_s_wait_tensorcnt(0);
    __syncthreads();
  }

#pragma unroll
  for (int mt = 0; mt < 2; ++mt) {
#pragma unroll
    for (int nt = 0; nt < 4; ++nt) {
      const int n  = bN + (wn * 4 + nt) * 16 + lr;
      const float bv = bias[n];
#pragma unroll
      for (int v = 0; v < 8; ++v) {
        const int m = bM + (wm * 2 + mt) * 16 + half * 8 + v;
        float x = acc[mt][nt][v] + bv;
        if (gelu) x = gelu_exact(x);
        const size_t idx = (size_t)m * DMODEL + n;
        if (outF) outF[idx] = x;
        if (outB) outB[idx] = f2bf(x);
      }
    }
  }
}

// ---------------------------------------------------------------------------
// Per-token head-mixing attention, one wave/token. 4 + 8 WMMAs per token.
// Fuses residual: R1 = X + y.
// ---------------------------------------------------------------------------
__global__ __launch_bounds__(256) void attn_headmix(
    const unsigned short* __restrict__ Q, const unsigned short* __restrict__ K,
    const unsigned short* __restrict__ V, const float* __restrict__ X,
    float* __restrict__ R1)
{
  __shared__ __align__(16) float          sc[8][16][16];
  __shared__ __align__(16) unsigned short attb[8][16][16];
  __shared__ __align__(16) unsigned short vT[8][128][16];

  const int tid  = threadIdx.x;
  const int lane = tid & 31;
  const int wid  = tid >> 5;
  const int lr   = lane & 15;
  const int half = lane >> 4;
  const int token = blockIdx.x * 8 + wid;
  const size_t base = (size_t)token * DMODEL;

  {  // transpose v[16][128] -> vT[128][16]
    const unsigned short* vrow = V + base + (size_t)lr * DHEAD + half * 64;
#pragma unroll
    for (int j = 0; j < 8; ++j) {
      union { uint4 q; unsigned short s[8]; } d;
      d.q = ((const uint4*)vrow)[j];
#pragma unroll
      for (int e = 0; e < 8; ++e) vT[wid][half * 64 + j * 8 + e][lr] = d.s[e];
    }
  }

  const v8f zero8 = {0.f, 0.f, 0.f, 0.f, 0.f, 0.f, 0.f, 0.f};
  v8f s = zero8;
#pragma unroll
  for (int kc = 0; kc < 4; ++kc) {
    Frag16 fq, fk;
    const unsigned short* qp = Q + base + (size_t)lr * DHEAD + kc * 32;
    fq.q[0] = *(const uint4*)(qp + half * 8);
    fq.q[1] = *(const uint4*)(qp + 16 + half * 8);
    const unsigned short* kp = K + base + (size_t)lr * DHEAD + kc * 32;
    fk.q[0] = *(const uint4*)(kp + half * 8);
    fk.q[1] = *(const uint4*)(kp + 16 + half * 8);
    s = wmma_bf16(fq.v, fk.v, s);
  }
#pragma unroll
  for (int v = 0; v < 8; ++v)
    sc[wid][half * 8 + v][lr] = s[v] * 0.08838834764831845f;  // 1/sqrt(128)

  __syncthreads();

  if (half == 0) {  // row softmax: lane lr owns row lr
    float r[16], m = -1e30f;
#pragma unroll
    for (int j = 0; j < 16; ++j) { r[j] = sc[wid][lr][j]; m = fmaxf(m, r[j]); }
    float sum = 0.f;
#pragma unroll
    for (int j = 0; j < 16; ++j) { r[j] = expf(r[j] - m); sum += r[j]; }
    const float inv = 1.0f / sum;
#pragma unroll
    for (int j = 0; j < 16; ++j) attb[wid][lr][j] = f2bf(r[j] * inv);
  }
  __syncthreads();

  Frag16 fat;
  fat.q[0] = *(const uint4*)&attb[wid][lr][half * 8];
  fat.q[1] = make_uint4(0, 0, 0, 0);     // K 16..31 zero padded

#pragma unroll
  for (int nt = 0; nt < 8; ++nt) {
    Frag16 fv;
    fv.q[0] = make_uint4(0, 0, 0, 0);
    fv.q[1] = make_uint4(0, 0, 0, 0);
    fv.q[0] = *(const uint4*)&vT[wid][nt * 16 + lr][half * 8];
    v8f y = wmma_bf16(fat.v, fv.v, zero8);
#pragma unroll
    for (int v = 0; v < 8; ++v) {
      const int h = half * 8 + v;
      const size_t idx = base + (size_t)h * DHEAD + nt * 16 + lr;
      R1[idx] = X[idx] + y[v];
    }
  }
}

// ---------------------------------------------------------------------------
// out = LayerNorm(a (+ b)) * g + beta; optional bf16 copy of the output.
// ---------------------------------------------------------------------------
__global__ __launch_bounds__(256) void ln_residual(
    const float* __restrict__ a, const float* __restrict__ b,
    const float* __restrict__ g, const float* __restrict__ beta,
    float* __restrict__ out, unsigned short* __restrict__ outB)
{
  __shared__ float s1[256];
  __shared__ float s2[256];
  const int tid = threadIdx.x;
  const size_t base = (size_t)blockIdx.x * DMODEL;

  float x[8]; float p1 = 0.f, p2 = 0.f;
#pragma unroll
  for (int j = 0; j < 8; ++j) {
    const int i = tid + j * 256;
    float v = a[base + i];
    if (b) v += b[base + i];
    x[j] = v; p1 += v; p2 += v * v;
  }
  s1[tid] = p1; s2[tid] = p2;
  __syncthreads();
  for (int off = 128; off > 0; off >>= 1) {
    if (tid < off) { s1[tid] += s1[tid + off]; s2[tid] += s2[tid + off]; }
    __syncthreads();
  }
  const float mean = s1[0] * (1.0f / DMODEL);
  const float var  = s2[0] * (1.0f / DMODEL) - mean * mean;
  const float inv  = rsqrtf(var + 1e-5f);
#pragma unroll
  for (int j = 0; j < 8; ++j) {
    const int i = tid + j * 256;
    const float o = (x[j] - mean) * inv * g[i] + beta[i];
    out[base + i] = o;
    if (outB) outB[base + i] = f2bf(o);
  }
}

// ---------------------------------------------------------------------------
extern "C" void kernel_launch(void* const* d_in, const int* in_sizes, int n_in,
                              void* d_out, int out_size, void* d_ws, size_t ws_size,
                              hipStream_t stream) {
  const float* X     = (const float*)d_in[0];
  const float* Wq    = (const float*)d_in[1];
  const float* bq    = (const float*)d_in[2];
  const float* Wk    = (const float*)d_in[3];
  const float* bk    = (const float*)d_in[4];
  const float* Wv    = (const float*)d_in[5];
  const float* bv    = (const float*)d_in[6];
  const float* g1    = (const float*)d_in[7];
  const float* beta1 = (const float*)d_in[8];
  const float* W1    = (const float*)d_in[9];
  const float* b1    = (const float*)d_in[10];
  const float* W2    = (const float*)d_in[11];
  const float* b2    = (const float*)d_in[12];
  const float* g2    = (const float*)d_in[13];
  const float* beta2 = (const float*)d_in[14];

  const size_t NEL = (size_t)NTOK * DMODEL;     // 33.5M
  const size_t WEL = (size_t)DMODEL * DMODEL;   // 4.2M
  char* p = (char*)d_ws;
  unsigned short* Xb  = (unsigned short*)p; p += NEL * 2;
  unsigned short* WqT = (unsigned short*)p; p += WEL * 2;
  unsigned short* WkT = (unsigned short*)p; p += WEL * 2;
  unsigned short* WvT = (unsigned short*)p; p += WEL * 2;
  unsigned short* W1T = (unsigned short*)p; p += WEL * 2;
  unsigned short* W2T = (unsigned short*)p; p += WEL * 2;
  unsigned short* Qb  = (unsigned short*)p; p += NEL * 2;
  unsigned short* Kb  = (unsigned short*)p; p += NEL * 2;
  unsigned short* Vb  = (unsigned short*)p; p += NEL * 2;
  float* R1 = (float*)p; p += NEL * 4;
  float* X1 = (float*)p; p += NEL * 4;
  unsigned short* X1b = (unsigned short*)p; p += NEL * 2;
  unsigned short* Hb  = Qb;   // FFN hidden reuses dead Q buffer
  float* Y2 = R1;             // FFN2 output reuses dead R1

  dim3 gG(DMODEL / 128, NTOK / 128);    // 16 x 128
  dim3 gT(DMODEL / 32, DMODEL / 32);    // 64 x 64

  f32_to_bf16    <<<NEL / 2048, 256, 0, stream>>>(X, Xb);
  w_transpose_bf16<<<gT, 256, 0, stream>>>(Wq, WqT);
  w_transpose_bf16<<<gT, 256, 0, stream>>>(Wk, WkT);
  w_transpose_bf16<<<gT, 256, 0, stream>>>(Wv, WvT);
  w_transpose_bf16<<<gT, 256, 0, stream>>>(W1, W1T);
  w_transpose_bf16<<<gT, 256, 0, stream>>>(W2, W2T);

  gemm_bf16_tdm<<<gG, 256, 0, stream>>>(Xb, WqT, bq, nullptr, Qb, 0);
  gemm_bf16_tdm<<<gG, 256, 0, stream>>>(Xb, WkT, bk, nullptr, Kb, 0);
  gemm_bf16_tdm<<<gG, 256, 0, stream>>>(Xb, WvT, bv, nullptr, Vb, 0);
  attn_headmix <<<NTOK / 8, 256, 0, stream>>>(Qb, Kb, Vb, X, R1);
  ln_residual  <<<NTOK, 256, 0, stream>>>(R1, nullptr, g1, beta1, X1, X1b);
  gemm_bf16_tdm<<<gG, 256, 0, stream>>>(X1b, W1T, b1, nullptr, Hb, 1);
  gemm_bf16_tdm<<<gG, 256, 0, stream>>>(Hb, W2T, b2, Y2, nullptr, 0);
  ln_residual  <<<NTOK, 256, 0, stream>>>(X1, Y2, g2, beta2, (float*)d_out, nullptr);
}
